// TorchRandomProject_29678224016022
// MI455X (gfx1250) — compile-verified, matrix operand order checked by reading, not run on hardware
//
#include <hip/hip_runtime.h>

// GEMM: out[4096,4096] = X[4096,2048] * M[4096,2048]^T  (fp32 in/out)
// f16 WMMA (v_wmma_f32_16x16x32_f16) with f32 accumulation; fp32 -> f16
// via single-instruction v_cvt_pk_f16_f32 (RTZ). 128x128 block tile,
// 8 waves (wave32), 32x64 per wave (2x4 accs), K-step 32, double-buffered
// LDS with 80B pitch => conflict-free ds_load_b128 fragment fetches.

#define B_DIM 4096
#define I_DIM 2048
#define O_DIM 4096
#define TILE  128
#define KB    32
#define NKT   (I_DIM / KB)        // 64 K-steps
#define LDS_PITCH 80              // bytes per 32-element f16 row (64 + 16 pad)

typedef __attribute__((ext_vector_type(16))) _Float16 v16h;
typedef __attribute__((ext_vector_type(2)))  __fp16   fp16x2;
typedef __attribute__((ext_vector_type(8)))  float    v8f;

union FragCast { v16h v; uint4 q[2]; };

__device__ __forceinline__ unsigned int pack2h(float a, float b) {
  // v_cvt_pk_f16_f32: one instruction packs two floats to two halves
  fp16x2 p = __builtin_amdgcn_cvt_pkrtz(a, b);
  return __builtin_bit_cast(unsigned int, p);
}

__global__ __launch_bounds__(256)
void rp_wmma_gemm(const float* __restrict__ X,
                  const float* __restrict__ W,
                  float* __restrict__ Out) {
  __shared__ __align__(16) unsigned char ldsA[2][TILE * LDS_PITCH];
  __shared__ __align__(16) unsigned char ldsB[2][TILE * LDS_PITCH];

  const int tid   = threadIdx.x;        // 0..255
  const int lane  = tid & 31;
  const int wave  = tid >> 5;           // 0..7
  const int waveM = wave & 3;           // 4 wave-rows of 32
  const int waveN = wave >> 2;          // 2 wave-cols of 64

  const int rowTile = blockIdx.y * TILE;   // B dimension
  const int colTile = blockIdx.x * TILE;   // O dimension

  // Global staging: 256 threads cover a 128x32 fp32 tile in 4 passes of
  // 32 rows; each thread owns one float4 chunk (K sub-range) per pass.
  const int ldRow   = tid >> 3;            // 0..31
  const int ldChunk = tid & 7;             // 0..7 (float4 chunk in K)

  // Per-lane element offsets (fit in 32 bits); K advance stays uniform so
  // loads lower to scalar-base + vector-offset with immediate offsets.
  const unsigned aLaneOff = (unsigned)(rowTile + ldRow) * I_DIM + ldChunk * 4u;
  const unsigned bLaneOff = (unsigned)(colTile + ldRow) * I_DIM + ldChunk * 4u;

  float4 aReg[4], bReg[4];
  v8f acc[2][4] = {};

  const int half16 = (lane >> 4) << 4;     // 0 or 16: K-group byte offset
  const int fragRowLane = lane & 15;

  auto loadGlobal = [&](int kt) {
    const float* xk = X + (unsigned)(kt * KB);   // uniform K advance
    const float* wk = W + (unsigned)(kt * KB);
#pragma unroll
    for (int p = 0; p < 4; ++p) {
      aReg[p] = *(const float4*)(xk + aLaneOff + (unsigned)(p * 32 * I_DIM));
      bReg[p] = *(const float4*)(wk + bLaneOff + (unsigned)(p * 32 * I_DIM));
    }
  };

  auto storeLds = [&](int buf) {
#pragma unroll
    for (int p = 0; p < 4; ++p) {
      const int r   = p * 32 + ldRow;
      const int off = r * LDS_PITCH + ldChunk * 8;
      uint2 pa, pb;
      pa.x = pack2h(aReg[p].x, aReg[p].y);
      pa.y = pack2h(aReg[p].z, aReg[p].w);
      pb.x = pack2h(bReg[p].x, bReg[p].y);
      pb.y = pack2h(bReg[p].z, bReg[p].w);
      *(uint2*)&ldsA[buf][off] = pa;
      *(uint2*)&ldsB[buf][off] = pb;
    }
  };

  auto compute = [&](int buf) {
    FragCast a[2], b[4];
#pragma unroll
    for (int mi = 0; mi < 2; ++mi) {
      const int r = waveM * 32 + mi * 16 + fragRowLane;
      const unsigned char* p = &ldsA[buf][r * LDS_PITCH + half16];
      a[mi].q[0] = *(const uint4*)(p);
      a[mi].q[1] = *(const uint4*)(p + 32);
    }
#pragma unroll
    for (int ni = 0; ni < 4; ++ni) {
      const int r = waveN * 64 + ni * 16 + fragRowLane;
      const unsigned char* p = &ldsB[buf][r * LDS_PITCH + half16];
      b[ni].q[0] = *(const uint4*)(p);
      b[ni].q[1] = *(const uint4*)(p + 32);
    }
#pragma unroll
    for (int mi = 0; mi < 2; ++mi)
#pragma unroll
      for (int ni = 0; ni < 4; ++ni)
        acc[mi][ni] = __builtin_amdgcn_wmma_f32_16x16x32_f16(
            /*neg_a=*/false, a[mi].v, /*neg_b=*/false, b[ni].v,
            /*c_mod=*/(short)0, acc[mi][ni],
            /*reuse_a=*/false, /*reuse_b=*/false);
  };

  // ---- pipeline: preload K-tile 0 ----
  loadGlobal(0);
  storeLds(0);
  __syncthreads();

  int buf = 0;
  for (int kt = 0; kt < NKT; ++kt) {
    if (kt + 2 < NKT) {
      // prefetch tile kt+2 into GL2 (global_prefetch_b8)
      const unsigned kp = (unsigned)((kt + 2) * KB);
      __builtin_prefetch((const void*)(X + kp + aLaneOff), 0, 3);
      __builtin_prefetch((const void*)(W + kp + bLaneOff), 0, 3);
    }
    if (kt + 1 < NKT) loadGlobal(kt + 1);   // long-latency loads overlap WMMA
    compute(buf);
    if (kt + 1 < NKT) {
      storeLds(buf ^ 1);
      __syncthreads();
      buf ^= 1;
    }
  }

  // ---- writeback: C/D layout: VGPR j -> row j (lanes 0-15) / j+8 (16-31) ----
  const int rowUpper = (lane >> 4) << 3;    // +8 for lanes 16..31
#pragma unroll
  for (int mi = 0; mi < 2; ++mi) {
#pragma unroll
    for (int ni = 0; ni < 4; ++ni) {
      const int col  = colTile + waveN * 64 + ni * 16 + fragRowLane;
      const int row0 = rowTile + waveM * 32 + mi * 16 + rowUpper;
#pragma unroll
      for (int j = 0; j < 8; ++j)
        Out[(size_t)(row0 + j) * O_DIM + col] = acc[mi][ni][j];
    }
  }
}

extern "C" void kernel_launch(void* const* d_in, const int* in_sizes, int n_in,
                              void* d_out, int out_size, void* d_ws, size_t ws_size,
                              hipStream_t stream) {
  (void)in_sizes; (void)n_in; (void)out_size; (void)d_ws; (void)ws_size;
  const float* X = (const float*)d_in[0];     // [4096, 2048]
  const float* W = (const float*)d_in[1];     // [1, 4096, 2048] -> rows of M
  float* Out = (float*)d_out;                 // [4096, 4096]

  dim3 grid(O_DIM / TILE, B_DIM / TILE);      // (32, 32)
  dim3 block(256);
  rp_wmma_gemm<<<grid, block, 0, stream>>>(X, W, Out);
}
